// decoder_only_lstm_32349693673760
// MI455X (gfx1250) — matrix operand order repaired and checked by reading, not compiled, not run on hardware
//
#include <hip/hip_runtime.h>
#include <hip/hip_bf16.h>
#include <math.h>

#define BB      128
#define LPAST   365
#define DIN     32
#define HHOR    8
#define NSTAT   27
#define HID     1024
#define TEMBN   256
#define TSTEPS  372       // (LPAST-1) past steps + HHOR future steps
#define KDIM    1088      // 64 (feats 60 padded to 64) + 1024 (hidden)
#define NGATES  4096      // 4*HID, interleaved: n = 4*j + gate
#define TILE_N  32        // gate columns per block -> 8 hidden units per block

typedef __attribute__((ext_vector_type(16))) __bf16 v16bf;
typedef __attribute__((ext_vector_type(8)))  float  v8f;

union Frag16 { v16bf v; uint4 u[2]; };

__device__ __forceinline__ unsigned short f2bf(float f) {
    unsigned int u = __float_as_uint(f);
    u += 0x7FFFu + ((u >> 16) & 1u);   // round to nearest even
    return (unsigned short)(u >> 16);
}
__device__ __forceinline__ float bf2f(unsigned short u) {
    return __uint_as_float(((unsigned int)u) << 16);
}
__device__ __forceinline__ float sigm(float x) { return 1.0f / (1.0f + __expf(-x)); }
__device__ __forceinline__ float tanh_fast(float x) {       // branch-free tanh
    float xx = fminf(fmaxf(x, -15.0f), 15.0f);
    float e  = __expf(2.0f * xx);
    return (e - 1.0f) / (e + 1.0f);
}

// ---- prep: interleaved bf16 weight matrix Wn[n][k], n = 4*j + g, k in [0,1088) ----
__global__ void k_build_w(const float* __restrict__ W_ih, const float* __restrict__ W_hh,
                          unsigned short* __restrict__ Wn) {
    int idx = blockIdx.x * blockDim.x + threadIdx.x;       // < 4096*1088
    int n = idx / KDIM, k = idx % KDIM;
    int j = n >> 2, g = n & 3;
    int r = g * HID + j;                                   // torch gate order i,f,g,o
    float v = 0.0f;
    if (k < 60)       v = W_ih[r * 60 + k];
    else if (k >= 64) v = W_hh[r * HID + (k - 64)];
    Wn[idx] = f2bf(v);
}

__global__ void k_bias4(const float* __restrict__ b_ih, const float* __restrict__ b_hh,
                        float* __restrict__ bias4) {
    int n = blockIdx.x * blockDim.x + threadIdx.x;         // < 4096
    int j = n >> 2, g = n & 3;
    bias4[n] = b_ih[g * HID + j] + b_hh[g * HID + j];
}

// ---- prep: ALL timesteps' feature columns, bf16, [t][b][64] ----
__global__ void k_feats(const float* __restrict__ x_past, const float* __restrict__ x_fut,
                        const float* __restrict__ noisy, const float* __restrict__ statA,
                        unsigned short* __restrict__ featsAll) {
    int idx = blockIdx.x * blockDim.x + threadIdx.x;       // < 372*128*64
    int k  = idx & 63;
    int bt = idx >> 6;
    int b  = bt & 127;
    int t  = bt >> 7;
    float v = 0.0f;
    if (k < DIN) {
        v = (t < LPAST) ? x_past[((size_t)b * LPAST + t) * DIN + k]
                        : x_fut[((size_t)b * (HHOR - 1) + (t - LPAST)) * DIN + k];
    } else if (k == DIN) {
        v = (t >= LPAST - 1) ? noisy[b * HHOR + (t - (LPAST - 1))] : 0.0f;
    } else if (k < 60) {
        v = statA[b * NSTAT + (k - 33)];
    }
    featsAll[idx] = f2bf(v);
}

// ---- per-step: WMMA GEMM (128 x 32 gate cols, K=1088) + LSTM cell update ----
// A operand is split: K<64 from feats_t [b][64], K>=64 from hin bf16 [b][1024].
// h ping-pong: reads hin, writes hout (full coverage across the 128 blocks).
__global__ __launch_bounds__(256)
void k_step(const unsigned short* __restrict__ feats_t,
            const unsigned short* __restrict__ hin,
            const unsigned short* __restrict__ Wn,
            const float* __restrict__ bias4,
            unsigned short* __restrict__ hout,
            float* __restrict__ c, float* __restrict__ fut) {
    __shared__ float gates[128 * TILE_N];                  // 16 KB
    const int tid  = threadIdx.x;
    const int wave = tid >> 5;
    const int lane = tid & 31;
    const int bn   = blockIdx.x;                           // 0..127
    const int nstrip = wave & 1;
    const int mt0    = (wave >> 1) * 2;                    // two 16-row M tiles per wave
    const int l15 = lane & 15;
    const int lhi = lane >> 4;

    // B fragment: lane holds col n, K = 8*lhi + {0..7} and {16..23} (N-major Wn)
    const int ncol = bn * TILE_N + nstrip * 16 + l15;
    const unsigned short* wrow = Wn + (size_t)ncol * KDIM + lhi * 8;

    // A fragment: lane holds row m, K = 8*lhi + {0..7} and {16..23}
    const int m0 = mt0 * 16 + l15;

    v8f c0 = {}; v8f c1 = {};

    // --- feats region: K tiles 0 and 32 (row stride 64) ---
    {
        const unsigned short* a0p = feats_t + (size_t)m0 * 64 + lhi * 8;
        const unsigned short* a1p = a0p + (size_t)16 * 64;
#pragma unroll
        for (int k0 = 0; k0 < 64; k0 += 32) {
            Frag16 a0, a1, bf;
            a0.u[0] = *(const uint4*)(a0p + k0);
            a0.u[1] = *(const uint4*)(a0p + k0 + 16);
            a1.u[0] = *(const uint4*)(a1p + k0);
            a1.u[1] = *(const uint4*)(a1p + k0 + 16);
            bf.u[0] = *(const uint4*)(wrow + k0);
            bf.u[1] = *(const uint4*)(wrow + k0 + 16);
            c0 = __builtin_amdgcn_wmma_f32_16x16x32_bf16(false, a0.v, false, bf.v,
                                                         (short)0, c0, false, false);
            c1 = __builtin_amdgcn_wmma_f32_16x16x32_bf16(false, a1.v, false, bf.v,
                                                         (short)0, c1, false, false);
        }
    }
    // --- hidden region: K tiles 64..1087 (row stride 1024) ---
    {
        const unsigned short* a0p = hin + (size_t)m0 * HID + lhi * 8;
        const unsigned short* a1p = a0p + (size_t)16 * HID;
        const unsigned short* wp  = wrow + 64;
        for (int k0 = 0; k0 < HID; k0 += 32) {
            Frag16 a0, a1, bf;
            a0.u[0] = *(const uint4*)(a0p + k0);
            a0.u[1] = *(const uint4*)(a0p + k0 + 16);
            a1.u[0] = *(const uint4*)(a1p + k0);
            a1.u[1] = *(const uint4*)(a1p + k0 + 16);
            bf.u[0] = *(const uint4*)(wp + k0);
            bf.u[1] = *(const uint4*)(wp + k0 + 16);
            c0 = __builtin_amdgcn_wmma_f32_16x16x32_bf16(false, a0.v, false, bf.v,
                                                         (short)0, c0, false, false);
            c1 = __builtin_amdgcn_wmma_f32_16x16x32_bf16(false, a1.v, false, bf.v,
                                                         (short)0, c1, false, false);
        }
    }

    // spill C tiles to LDS: lane holds col l15, rows 8*lhi + v
    const int colLoc = nstrip * 16 + l15;
#pragma unroll
    for (int v = 0; v < 8; ++v) {
        int mA = mt0 * 16 + lhi * 8 + v;
        gates[mA * TILE_N + colLoc]        = c0[v];
        gates[(mA + 16) * TILE_N + colLoc] = c1[v];
    }
    __syncthreads();

    // cell update: block owns 8 hidden units x 128 batch = 1024 outputs
    const float* bias = bias4 + bn * TILE_N;
#pragma unroll
    for (int r = 0; r < 4; ++r) {
        int idx = tid + r * 256;                           // 0..1023
        int b  = idx >> 3;
        int jj = idx & 7;
        float4 g4 = *(const float4*)&gates[b * TILE_N + 4 * jj];
        float gi = g4.x + bias[4 * jj + 0];
        float gf = g4.y + bias[4 * jj + 1];
        float gg = g4.z + bias[4 * jj + 2];
        float go = g4.w + bias[4 * jj + 3];
        int j = bn * 8 + jj;
        size_t hidx = (size_t)b * HID + j;
        float cn = sigm(gf) * c[hidx] + sigm(gi) * tanh_fast(gg);
        float hn = sigm(go) * tanh_fast(cn);
        c[hidx]    = cn;
        hout[hidx] = f2bf(hn);
        if (fut) fut[(size_t)b * (HHOR * HID) + j] = hn;
    }
}

// ---- time embedding: MPFourier -> Linear -> SiLU -> Linear, added into h(bf16) and c ----
__global__ void k_emb(const float* __restrict__ t, const float* __restrict__ freqs,
                      const float* __restrict__ phases, float* __restrict__ emb) {
    int b = blockIdx.x, e = threadIdx.x;                   // 128 x 256
    emb[b * TEMBN + e] = __cosf(t[b] * freqs[e] + phases[e]) * 1.41421356237309515f;
}

__global__ void k_mlp1(const float* __restrict__ emb, const float* __restrict__ W1,
                       const float* __restrict__ b1, float* __restrict__ hmid) {
    int b = blockIdx.x, o = threadIdx.x;                   // 128 x 512
    const float* w = W1 + (size_t)o * TEMBN;
    const float* e = emb + (size_t)b * TEMBN;
    float acc = b1[o];
    for (int k = 0; k < TEMBN; ++k) acc += e[k] * w[k];
    hmid[b * (2 * TEMBN) + o] = acc * sigm(acc);           // SiLU
}

__global__ void k_mlp2_add(const float* __restrict__ hmid, const float* __restrict__ W2,
                           const float* __restrict__ b2, unsigned short* __restrict__ hbf,
                           float* __restrict__ c) {
    int b = blockIdx.x, o = blockIdx.y * 256 + threadIdx.x; // (128,4) x 256
    const float* w  = W2 + (size_t)o * (2 * TEMBN);
    const float* hm = hmid + (size_t)b * (2 * TEMBN);
    float acc = b2[o];
    for (int k = 0; k < 2 * TEMBN; ++k) acc += hm[k] * w[k];
    size_t idx = (size_t)b * HID + o;
    hbf[idx] = f2bf(bf2f(hbf[idx]) + acc);
    c[idx] += acc;
}

// ---- head: out[b,s] = fut_h[b,s,:] . Wh + bh ----
__global__ void k_head(const float* __restrict__ futh, const float* __restrict__ Wh,
                       const float* __restrict__ bh, float* __restrict__ out) {
    __shared__ float red[256];
    int bs = blockIdx.x;                                   // b*8 + s, 1024 blocks
    const float* row = futh + (size_t)bs * HID;
    float acc = 0.0f;
    for (int k = threadIdx.x; k < HID; k += 256) acc += row[k] * Wh[k];
    red[threadIdx.x] = acc;
    __syncthreads();
    for (int s = 128; s > 0; s >>= 1) {
        if (threadIdx.x < s) red[threadIdx.x] += red[threadIdx.x + s];
        __syncthreads();
    }
    if (threadIdx.x == 0) out[bs] = red[0] + bh[0];
}

extern "C" void kernel_launch(void* const* d_in, const int* in_sizes, int n_in,
                              void* d_out, int out_size, void* d_ws, size_t ws_size,
                              hipStream_t stream) {
    const float* x_past = (const float*)d_in[0];
    const float* noisy  = (const float*)d_in[1];
    const float* tt     = (const float*)d_in[2];
    const float* x_fut  = (const float*)d_in[3];
    const float* statA  = (const float*)d_in[4];
    const float* W_ih   = (const float*)d_in[5];
    const float* W_hh   = (const float*)d_in[6];
    const float* b_ih   = (const float*)d_in[7];
    const float* b_hh   = (const float*)d_in[8];
    const float* freqs  = (const float*)d_in[9];
    const float* phases = (const float*)d_in[10];
    const float* W1     = (const float*)d_in[11];
    const float* b1     = (const float*)d_in[12];
    const float* W2     = (const float*)d_in[13];
    const float* b2     = (const float*)d_in[14];
    const float* Wh     = (const float*)d_in[15];
    const float* bh     = (const float*)d_in[16];

    char* ws = (char*)d_ws;
    size_t off = 0;
    auto alloc = [&](size_t bytes) {
        void* p = ws + off;
        off += (bytes + 255) & ~(size_t)255;
        return p;
    };
    unsigned short* Wn       = (unsigned short*)alloc((size_t)NGATES * KDIM * 2);   // 8.9 MB
    unsigned short* featsAll = (unsigned short*)alloc((size_t)TSTEPS * BB * 64 * 2);// 6.1 MB
    unsigned short* hbf0     = (unsigned short*)alloc((size_t)BB * HID * 2);
    unsigned short* hbf1     = (unsigned short*)alloc((size_t)BB * HID * 2);
    float* bias4 = (float*)alloc((size_t)NGATES * 4);
    float* c     = (float*)alloc((size_t)BB * HID * 4);
    float* futh  = (float*)alloc((size_t)BB * HHOR * HID * 4);                      // 4 MB
    float* emb   = (float*)alloc((size_t)BB * TEMBN * 4);
    float* hmid  = (float*)alloc((size_t)BB * 2 * TEMBN * 4);

    unsigned short* hbf[2] = { hbf0, hbf1 };

    hipMemsetAsync(hbf0, 0, (size_t)BB * HID * 2, stream);
    hipMemsetAsync(c,    0, (size_t)BB * HID * 4, stream);

    k_build_w<<<(NGATES * KDIM) / 256, 256, 0, stream>>>(W_ih, W_hh, Wn);
    k_bias4<<<NGATES / 256, 256, 0, stream>>>(b_ih, b_hh, bias4);
    k_feats<<<(TSTEPS * BB * 64) / 256, 256, 0, stream>>>(x_past, x_fut, noisy, statA, featsAll);

    // past pass: t = 0 .. L-2 (364 steps); h ping-pongs by parity
    for (int t = 0; t < LPAST - 1; ++t) {
        k_step<<<NGATES / TILE_N, 256, 0, stream>>>(
            featsAll + (size_t)t * BB * 64, hbf[t & 1], Wn, bias4,
            hbf[(t + 1) & 1], c, nullptr);
    }

    // time-embedding bottleneck, added into current h (in hbf[0] after 364 steps) and c
    k_emb<<<BB, TEMBN, 0, stream>>>(tt, freqs, phases, emb);
    k_mlp1<<<BB, 2 * TEMBN, 0, stream>>>(emb, W1, b1, hmid);
    k_mlp2_add<<<dim3(BB, 4), 256, 0, stream>>>(hmid, W2, b2, hbf[(LPAST - 1) & 1], c);

    // future pass: t = L-1 .. L-1+H-1 (8 steps), record h into futh
    for (int s = 0; s < HHOR; ++s) {
        int t = (LPAST - 1) + s;
        k_step<<<NGATES / TILE_N, 256, 0, stream>>>(
            featsAll + (size_t)t * BB * 64, hbf[t & 1], Wn, bias4,
            hbf[(t + 1) & 1], c, futh + (size_t)s * HID);
    }

    k_head<<<BB * HHOR, 256, 0, stream>>>(futh, Wh, bh, (float*)d_out);
}
